// Seq2Seq_41420664603256
// MI455X (gfx1250) — compile-verified, matrix operand order checked by reading, not run on hardware
//
#include <hip/hip_runtime.h>

// ---------------- problem constants (from reference) ----------------
#define BATCH 128
#define TSEQ  48
#define EMB   40
#define HID   256
#define GATE  768     // 3*HID
#define ATTH  200
#define READN 100
#define FF1N  100
#define VOCAB 30000
#define CCAT  512     // 2*HID
// padded (K -> mult of 32, N -> mult of 32 so chains stay aligned)
#define EMBP  64
#define ATTP  224
#define RDP   128
#define FF1P  128
#define VOCP  30016   // 30000 -> 938*32

typedef __attribute__((ext_vector_type(16))) __bf16 v16bf;
typedef __attribute__((ext_vector_type(8)))  __bf16 v8bf;
typedef __attribute__((ext_vector_type(8)))  float  v8f;

#define DEV_INLINE __device__ __forceinline__

// ---------------- staging kernels (run once per launch) ----------------
__global__ void zero_f32(float* __restrict__ p, int n) {
  int i = blockIdx.x * blockDim.x + threadIdx.x;
  if (i < n) p[i] = 0.0f;
}

// W:[K,N] f32 row-major -> Wt:[Npad,Kpad] bf16, transposed, zero pad
__global__ void stage_wt(const float* __restrict__ W, __bf16* __restrict__ Wt,
                         int K, int N, int Kpad, int Npad) {
  int i = blockIdx.x * blockDim.x + threadIdx.x;
  if (i >= Npad * Kpad) return;
  int n = i / Kpad, k = i % Kpad;
  Wt[i] = (n < N && k < K) ? (__bf16)W[(long)k * N + n] : (__bf16)0.f;
}

// W already [N,K] -> Wb:[Npad,Kpad] bf16, zero pad
__global__ void stage_w(const float* __restrict__ W, __bf16* __restrict__ Wb,
                        int K, int N, int Kpad, int Npad) {
  int i = blockIdx.x * blockDim.x + threadIdx.x;
  if (i >= Npad * Kpad) return;
  int n = i / Kpad, k = i % Kpad;
  Wb[i] = (n < N && k < K) ? (__bf16)W[(long)n * K + k] : (__bf16)0.f;
}

__global__ void stage_bias(const float* __restrict__ b, float* __restrict__ bp,
                           int N, int Npad) {
  int i = blockIdx.x * blockDim.x + threadIdx.x;
  if (i < Npad) bp[i] = (i < N) ? b[i] : 0.0f;
}

// A[(t*BATCH + b)*EMBP + e] = emb[tok[b*TSEQ + t]*EMB + e], pad cols zeroed
__global__ void embed_gather(const int* __restrict__ tok,
                             const float* __restrict__ emb,
                             float* __restrict__ A) {
  int i = blockIdx.x * blockDim.x + threadIdx.x;
  if (i >= TSEQ * BATCH * EMBP) return;
  int e  = i % EMBP;
  int bt = i / EMBP;
  int b  = bt % BATCH;
  int t  = bt / BATCH;
  A[i] = (e < EMB) ? emb[(long)tok[b * TSEQ + t] * EMB + e] : 0.0f;
}

// ---------------- WMMA bf16 GEMM (check-free hot loop) ----------------
//   C[M,Npad] = act(A[M,Kpad(f32)] * Wt[Npad,Kpad(bf16)]^T + bias)
// One wave per 16x16 C tile; M mult of 16, Npad mult of 16, Kpad mult of 32.
// act: 0=none 1=relu 2=tanh
__global__ void gemm_bf16_wmma(const float* __restrict__ A, int lda,
                               const __bf16* __restrict__ Wt,
                               const float* __restrict__ bias,
                               float* __restrict__ C, int ldc,
                               int M, int Npad, int Kpad, int act) {
  const int lane   = threadIdx.x & 31;
  const int wave   = threadIdx.x >> 5;
  const int tilesN = Npad >> 4;
  const long tile  = (long)blockIdx.x * (blockDim.x >> 5) + wave;
  if (tile >= (long)(M >> 4) * tilesN) return;   // wave-uniform: EXEC all-1s at wmma
  const int tm = (int)(tile / tilesN);
  const int tn = (int)(tile % tilesN);
  const int half = lane >> 4;                    // K-half select (ISA A/B layout)
  const int r    = lane & 15;

  // lane's A row / B column base, offset to this half's first K block
  const float*  __restrict__ Ar = A  + (long)(tm * 16 + r) * lda        + (half << 3);
  const __bf16* __restrict__ Br = Wt + (long)(tn * 16 + r) * (long)Kpad + (half << 3);

  v8f acc = {};
  for (int k0 = 0; k0 < Kpad; k0 += 32) {
    // A: 8 consecutive f32 at k0, 8 more at k0+16 (32B aligned)
    v8f fa0 = *(const v8f*)(Ar + k0);
    v8f fa1 = *(const v8f*)(Ar + k0 + 16);
    // B: 8 consecutive bf16 at k0, 8 more at k0+16 (16B aligned)
    v8bf bb0 = *(const v8bf*)(Br + k0);
    v8bf bb1 = *(const v8bf*)(Br + k0 + 16);
    v16bf af = __builtin_shufflevector(__builtin_convertvector(fa0, v8bf),
                                       __builtin_convertvector(fa1, v8bf),
                                       0,1,2,3,4,5,6,7,8,9,10,11,12,13,14,15);
    v16bf bf = __builtin_shufflevector(bb0, bb1,
                                       0,1,2,3,4,5,6,7,8,9,10,11,12,13,14,15);
    acc = __builtin_amdgcn_wmma_f32_16x16x32_bf16(false, af, false, bf,
                                                  (short)0, acc, false, false);
  }

  const int n = tn * 16 + r;
  const float bv = bias ? bias[n] : 0.0f;
  float* Cp = C + (long)(tm * 16 + (half << 3)) * ldc + n;  // C/D: M = rr + 8*half
#pragma unroll
  for (int rr = 0; rr < 8; ++rr) {
    float v = acc[rr] + bv;
    if (act == 1)      v = fmaxf(v, 0.0f);
    else if (act == 2) v = tanhf(v);
    Cp[(long)rr * ldc] = v;
  }
}

// ---------------- GRU gate fusion (Keras reset_after=True) ----------------
DEV_INLINE float sigm(float x) { return 1.0f / (1.0f + __expf(-x)); }

__global__ void gru_elem(const float* __restrict__ gi,   // [B,3H] = x@Wi + b0
                         const float* __restrict__ gh,   // [B,3H] = h@Wr + b1
                         const float* __restrict__ h_in, // [B,H]
                         float* __restrict__ h_out) {    // [B,H]
  int i = blockIdx.x * blockDim.x + threadIdx.x;
  if (i >= BATCH * HID) return;
  int b = i / HID, j = i % HID;
  const float* gib = gi + (long)b * GATE;
  const float* ghb = gh + (long)b * GATE;
  float z  = sigm(gib[j]            + ghb[j]);
  float rg = sigm(gib[HID + j]      + ghb[HID + j]);
  float nn = tanhf(gib[2 * HID + j] + rg * ghb[2 * HID + j]);
  float h  = h_in[i];
  h_out[i] = z * h + (1.0f - z) * nn;
}

// ---------------- dot-product attention, writes [state||ctx] ----------------
__global__ void attention_kernel(const float* __restrict__ state, // [B,H]
                                 const float* __restrict__ enc,   // [T,B,H]
                                 float* __restrict__ cc) {        // [B,2H]
  __shared__ float sh_state[HID];
  __shared__ float sh_sc[TSEQ];
  __shared__ float sh_inv;
  const int b = blockIdx.x;
  const int h = threadIdx.x;                                      // 256 threads
  sh_state[h] = state[(long)b * HID + h];
  __syncthreads();
  if (h < TSEQ) {
    const float* ep = enc + ((long)h * BATCH + b) * HID;
    float s = 0.0f;
    for (int k = 0; k < HID; ++k) s += sh_state[k] * ep[k];
    sh_sc[h] = s;
  }
  __syncthreads();
  if (h == 0) {
    float mx = -3.4e38f;
    for (int t = 0; t < TSEQ; ++t) mx = fmaxf(mx, sh_sc[t]);
    float sum = 0.0f;
    for (int t = 0; t < TSEQ; ++t) { float e = __expf(sh_sc[t] - mx); sh_sc[t] = e; sum += e; }
    sh_inv = 1.0f / sum;
  }
  __syncthreads();
  float ctx = 0.0f;
  for (int t = 0; t < TSEQ; ++t) ctx += sh_sc[t] * enc[((long)t * BATCH + b) * HID + h];
  cc[(long)b * CCAT + h]       = sh_state[h];
  cc[(long)b * CCAT + HID + h] = ctx * sh_inv;
}

// ---------------- vocab softmax: per-row stats then normalized store ----------------
__global__ void row_softmax_stats(const float* __restrict__ logits, // [B,VOCP]
                                  float* __restrict__ rmax, float* __restrict__ rsum) {
  __shared__ float red[256];
  const int b = blockIdx.x, tid = threadIdx.x;
  const float* row = logits + (long)b * VOCP;
  float mx = -3.4e38f;
  for (int v = tid; v < VOCAB; v += 256) mx = fmaxf(mx, row[v]);
  red[tid] = mx; __syncthreads();
  for (int s = 128; s > 0; s >>= 1) { if (tid < s) red[tid] = fmaxf(red[tid], red[tid + s]); __syncthreads(); }
  mx = red[0]; __syncthreads();
  float sum = 0.0f;
  for (int v = tid; v < VOCAB; v += 256) sum += __expf(row[v] - mx);
  red[tid] = sum; __syncthreads();
  for (int s = 128; s > 0; s >>= 1) { if (tid < s) red[tid] += red[tid + s]; __syncthreads(); }
  if (tid == 0) { rmax[b] = mx; rsum[b] = red[0]; }
}

__global__ void softmax_write(const float* __restrict__ logits,  // [B,VOCP]
                              const float* __restrict__ rmax,
                              const float* __restrict__ rsum,
                              float* __restrict__ out, int step) {
  int i = blockIdx.x * blockDim.x + threadIdx.x;
  if (i >= BATCH * VOCAB) return;
  int b = i / VOCAB, v = i % VOCAB;
  out[((long)b * TSEQ + step) * VOCAB + v] =
      __expf(logits[(long)b * VOCP + v] - rmax[b]) / rsum[b];
}

// ---------------- host side ----------------
static inline dim3 grid1(long n, int bs) { return dim3((unsigned)((n + bs - 1) / bs)); }

static void gemm(const float* A, int lda, const __bf16* Wt, const float* bias,
                 float* C, int ldc, int M, int Npad, int Kpad, int act, hipStream_t s) {
  long tiles  = (long)(M / 16) * (Npad / 16);
  long blocks = (tiles + 3) / 4;                       // 4 waves (16x16 tiles) / block
  gemm_bf16_wmma<<<dim3((unsigned)blocks), dim3(128), 0, s>>>(A, lda, Wt, bias, C, ldc,
                                                              M, Npad, Kpad, act);
}

extern "C" void kernel_launch(void* const* d_in, const int* in_sizes, int n_in,
                              void* d_out, int out_size, void* d_ws, size_t ws_size,
                              hipStream_t stream) {
  (void)in_sizes; (void)n_in; (void)out_size; (void)ws_size;
  const int*   enc_tok = (const int*)d_in[0];
  /* d_in[1] decoder_input: unused by the reference model */
  const float* fr_emb  = (const float*)d_in[2];
  const float* enc_Wi  = (const float*)d_in[3];
  const float* enc_Wr  = (const float*)d_in[4];
  const float* enc_b   = (const float*)d_in[5];
  const float* dec_Wi  = (const float*)d_in[6];
  const float* dec_Wr  = (const float*)d_in[7];
  const float* dec_b   = (const float*)d_in[8];
  const float* att_W1  = (const float*)d_in[9];
  const float* att_W2  = (const float*)d_in[10];
  const float* ff1_W   = (const float*)d_in[11];
  const float* ff1_b   = (const float*)d_in[12];
  const float* ff2_W   = (const float*)d_in[13];
  const float* ff2_b   = (const float*)d_in[14];
  float* out = (float*)d_out;

  // ---- workspace carve-up (~52 MB, L2-resident) ----
  char* w = (char*)d_ws;
  size_t off = 0;
  auto take = [&](size_t bytes) -> char* {
    char* p = w + off;
    off = (off + bytes + 255) & ~(size_t)255;
    return p;
  };
  __bf16* encWi_t = (__bf16*)take((size_t)GATE * EMBP * 2);   // [768,64]
  __bf16* encWr_t = (__bf16*)take((size_t)GATE * HID  * 2);   // [768,256]
  __bf16* decWi_t = (__bf16*)take((size_t)GATE * RDP  * 2);   // [768,128]
  __bf16* decWr_t = (__bf16*)take((size_t)GATE * HID  * 2);   // [768,256]
  __bf16* attW1_b = (__bf16*)take((size_t)ATTP * CCAT * 2);   // [224,512]
  __bf16* attW2_b = (__bf16*)take((size_t)RDP  * ATTP * 2);   // [128,224]
  __bf16* ff1_t   = (__bf16*)take((size_t)FF1P * HID  * 2);   // [128,256]
  __bf16* ff2_t   = (__bf16*)take((size_t)VOCP * FF1P * 2);   // [30016,128]
  float* ff1b_p  = (float*)take((size_t)FF1P * 4);
  float* ff2b_p  = (float*)take((size_t)VOCP * 4);
  float* embA    = (float*)take((size_t)TSEQ * BATCH * EMBP * 4);  // [T*B,64]
  float* enc_gi  = (float*)take((size_t)TSEQ * BATCH * GATE * 4);  // [T*B,768]
  float* enc_out = (float*)take((size_t)TSEQ * BATCH * HID  * 4);  // [T,B,256]
  float* h0      = (float*)take((size_t)BATCH * HID  * 4);
  float* gh      = (float*)take((size_t)BATCH * GATE * 4);
  float* stateA  = (float*)take((size_t)BATCH * HID  * 4);
  float* stateB  = (float*)take((size_t)BATCH * HID  * 4);
  float* concatB = (float*)take((size_t)BATCH * CCAT * 4);
  float* attHid  = (float*)take((size_t)BATCH * ATTP * 4);
  float* readB   = (float*)take((size_t)BATCH * RDP  * 4);
  float* dec_gi  = (float*)take((size_t)BATCH * GATE * 4);
  float* ff1o    = (float*)take((size_t)BATCH * FF1P * 4);
  float* logits  = (float*)take((size_t)BATCH * VOCP * 4);
  float* rmax    = (float*)take((size_t)BATCH * 4);
  float* rsum    = (float*)take((size_t)BATCH * 4);

  // ---- stage weights as transposed, zero-padded bf16 ----
  stage_wt<<<grid1((long)GATE * EMBP, 256), 256, 0, stream>>>(enc_Wi, encWi_t, EMB,   GATE, EMBP, GATE);
  stage_wt<<<grid1((long)GATE * HID,  256), 256, 0, stream>>>(enc_Wr, encWr_t, HID,   GATE, HID,  GATE);
  stage_wt<<<grid1((long)GATE * RDP,  256), 256, 0, stream>>>(dec_Wi, decWi_t, READN, GATE, RDP,  GATE);
  stage_wt<<<grid1((long)GATE * HID,  256), 256, 0, stream>>>(dec_Wr, decWr_t, HID,   GATE, HID,  GATE);
  stage_w <<<grid1((long)ATTP * CCAT, 256), 256, 0, stream>>>(att_W1, attW1_b, CCAT,  ATTH, CCAT, ATTP);
  stage_w <<<grid1((long)RDP  * ATTP, 256), 256, 0, stream>>>(att_W2, attW2_b, ATTH,  READN,ATTP, RDP);
  stage_wt<<<grid1((long)FF1P * HID,  256), 256, 0, stream>>>(ff1_W,  ff1_t,   HID,   FF1N, HID,  FF1P);
  stage_wt<<<grid1((long)VOCP * FF1P, 256), 256, 0, stream>>>(ff2_W,  ff2_t,   FF1N,  VOCAB,FF1P, VOCP);
  stage_bias<<<grid1(FF1P, 128), 128, 0, stream>>>(ff1_b, ff1b_p, FF1N,  FF1P);
  stage_bias<<<grid1(VOCP, 256), 256, 0, stream>>>(ff2_b, ff2b_p, VOCAB, VOCP);

  // ---- encoder: hoisted input projection, then sequential recurrent scan ----
  embed_gather<<<grid1((long)TSEQ * BATCH * EMBP, 256), 256, 0, stream>>>(enc_tok, fr_emb, embA);
  gemm(embA, EMBP, encWi_t, enc_b, enc_gi, GATE, TSEQ * BATCH, GATE, EMBP, 0, stream);
  zero_f32<<<grid1(BATCH * HID, 256), 256, 0, stream>>>(h0, BATCH * HID);

  const float* hprev = h0;
  for (int t = 0; t < TSEQ; ++t) {
    gemm(hprev, HID, encWr_t, enc_b + GATE, gh, GATE, BATCH, GATE, HID, 0, stream);
    float* hnext = enc_out + (long)t * BATCH * HID;
    gru_elem<<<grid1(BATCH * HID, 256), 256, 0, stream>>>(
        enc_gi + (long)t * BATCH * GATE, gh, hprev, hnext);
    hprev = hnext;
  }

  // ---- decoder scan ----
  const float* st = hprev;                  // encoder final state
  float* stbuf[2] = { stateA, stateB };
  for (int t = 0; t < TSEQ; ++t) {
    attention_kernel<<<dim3(BATCH), dim3(HID), 0, stream>>>(st, enc_out, concatB);
    gemm(concatB, CCAT, attW1_b, nullptr, attHid, ATTP, BATCH, ATTP, CCAT, 2, stream); // tanh
    gemm(attHid, ATTP, attW2_b, nullptr, readB, RDP, BATCH, RDP, ATTP, 0, stream);
    gemm(readB, RDP, decWi_t, dec_b,        dec_gi, GATE, BATCH, GATE, RDP, 0, stream);
    gemm(st,   HID,  decWr_t, dec_b + GATE, gh,     GATE, BATCH, GATE, HID, 0, stream);
    float* stn = stbuf[t & 1];
    gru_elem<<<grid1(BATCH * HID, 256), 256, 0, stream>>>(dec_gi, gh, st, stn);
    gemm(stn,  HID,  ff1_t, ff1b_p, ff1o,   FF1P, BATCH, FF1P, HID,  1, stream);       // relu
    gemm(ff1o, FF1P, ff2_t, ff2b_p, logits, VOCP, BATCH, VOCP, FF1P, 0, stream);       // big one
    row_softmax_stats<<<dim3(BATCH), dim3(256), 0, stream>>>(logits, rmax, rsum);
    softmax_write<<<grid1((long)BATCH * VOCAB, 256), 256, 0, stream>>>(logits, rmax, rsum, out, t);
    st = stn;
  }
}
